// MIMNCell_76613626626668
// MI455X (gfx1250) — matrix operand order repaired and checked by reading, not compiled, not run on hardware
//
#include <hip/hip_runtime.h>
#include <hip/hip_bf16.h>
#include <math.h>

// MIMN cell for MI455X (gfx1250, wave32).
// Kernel 0: one-shot f32->f16 weight conversion into workspace (L2-resident).
// Kernel 1: GRU + Wc projection via V_WMMA_F32_16X16X32_F16 (f16 operands, f32 accum).
// Kernel 2: cosine addressing + memory update, bandwidth-bound streaming in f32,
//           with global_prefetch_b8 to prime the sum_aggre stream.

typedef __attribute__((ext_vector_type(16))) _Float16 v16h;
typedef __attribute__((ext_vector_type(8)))  float    v8f;

constexpr int kEMB  = 64;
constexpr int kDIM  = 64;
constexpr int kMEM  = 64;
constexpr int kCTRL = 128;
constexpr int kTOT  = 268;   // NPH*NHEADS + 2*DIM = 140 + 128

__device__ __forceinline__ float sigmoidf_(float x) { return 1.f / (1.f + expf(-x)); }

__device__ __forceinline__ v8f wmma_f16(v16h a, v16h b, v8f c) {
  // 8 args: (neg_a, A, neg_b, B, c_mod, C, reuse_a, reuse_b)
  return __builtin_amdgcn_wmma_f32_16x16x32_f16(false, a, false, b, (short)0, c, false, false);
}

// ---- fragment loaders (ISA 7.12.2 layouts, wave32) -------------------------
// A (16x32 f16): lane = {m=lane&15, hi=lane>>4}; elem e -> K = ((e&8)<<1) + hi*8 + (e&7)
// B (32x16 f16): lane = {n=lane&15, hi=lane>>4}; elem e -> K = ((e&8)<<1) + hi*8 + (e&7)
// C/D (16x16 f32): vgpr r -> m = r + hi*8, n = lane&15

__device__ __forceinline__ v16h loadA_ctrl(const float* __restrict__ x,
                                           const float* __restrict__ pr,
                                           int rowbase, int kbase, int lane) {
  const int m = lane & 15, hi = lane >> 4;
  const int row = rowbase + m;
  v16h a;
#pragma unroll
  for (int e = 0; e < 16; ++e) {
    const int k = kbase + ((e & 8) << 1) + hi * 8 + (e & 7);
    const float v = (k < kEMB) ? x[(size_t)row * kEMB + k]
                               : pr[(size_t)row * kDIM + (k - kEMB)];
    a[e] = (_Float16)v;
  }
  return a;
}

__device__ __forceinline__ v16h loadA_g(const float* __restrict__ P, int ld,
                                        int rowbase, int kbase, int lane) {
  const int m = lane & 15, hi = lane >> 4;
  v16h a;
#pragma unroll
  for (int e = 0; e < 16; ++e) {
    const int k = kbase + ((e & 8) << 1) + hi * 8 + (e & 7);
    a[e] = (_Float16)P[(size_t)(rowbase + m) * ld + k];
  }
  return a;
}

__device__ __forceinline__ v16h loadA_lds(const _Float16* __restrict__ sH,
                                          int kbase, int lane) {
  const int m = lane & 15, hi = lane >> 4;
  v16h a;
#pragma unroll
  for (int e = 0; e < 16; ++e) {
    const int k = kbase + ((e & 8) << 1) + hi * 8 + (e & 7);
    a[e] = sH[m * kCTRL + k];
  }
  return a;
}

// B[k][n] = W[colbase+n][kbase+k] from pre-converted f16 weights (row-major,
// ldk columns). k = kbase + ((e&8)<<1) + hi*8 + (e&7) (matches A layout).
__device__ __forceinline__ v16h loadB_T(const _Float16* __restrict__ W, int ldk,
                                        int colbase, int colmax, int kbase, int lane) {
  const int n = lane & 15, hi = lane >> 4;
  const int col = colbase + n;
  const bool ok = (col < colmax);
  v16h b;
#pragma unroll
  for (int e = 0; e < 16; ++e) {
    const int k = kbase + ((e & 8) << 1) + hi * 8 + (e & 7);
    b[e] = ok ? W[(size_t)col * ldk + k] : (_Float16)0.f;
  }
  return b;
}

// ---------------------------------------------------------------------------
// Kernel 0: one-shot weight conversion f32 -> f16 (weights are shared by all
// 512 GEMM blocks; converting once removes ~16 v_cvt per WMMA from kernel 1).
// ---------------------------------------------------------------------------
__global__ __launch_bounds__(256) void k_prep(
    const float* __restrict__ W_ih, const float* __restrict__ W_hh,
    const float* __restrict__ Wc,
    _Float16* __restrict__ fWih, _Float16* __restrict__ fWhh,
    _Float16* __restrict__ fWc) {
  const int i = blockIdx.x * 256 + threadIdx.x;
  if (i < 384 * kCTRL) {
    fWih[i] = (_Float16)W_ih[i];
    fWhh[i] = (_Float16)W_hh[i];
  }
  if (i < kTOT * kCTRL) fWc[i] = (_Float16)Wc[i];
}

// ---------------------------------------------------------------------------
// Kernel 1: per-wave tile of 16 batch rows.
//   gi = [x,prev_read] @ W_ih^T ; gh = h_prev @ W_hh^T ; GRU fuse -> h
//   params = clip(h @ Wc^T + bc) -> k(tanh), beta(softplus), erase(sig), add(tanh)
// ---------------------------------------------------------------------------
__global__ __launch_bounds__(32) void k_controller(
    const float* __restrict__ x, const float* __restrict__ prev_read,
    const float* __restrict__ cs,
    const _Float16* __restrict__ W_ih, const _Float16* __restrict__ W_hh,
    const float* __restrict__ b_ih, const float* __restrict__ b_hh,
    const _Float16* __restrict__ Wc, const float* __restrict__ bc,
    float* __restrict__ o_h,
    float* __restrict__ wsK, float* __restrict__ wsBeta,
    float* __restrict__ wsE, float* __restrict__ wsA) {
  __shared__ _Float16 sH[16 * kCTRL];

  const int lane = threadIdx.x & 31;
  const int n = lane & 15, hi = lane >> 4;
  const int rowbase = blockIdx.x * 16;

  v16h aC[4], aH[4];
#pragma unroll
  for (int kk = 0; kk < 4; ++kk) {
    aC[kk] = loadA_ctrl(x, prev_read, rowbase, kk * 32, lane);
    aH[kk] = loadA_g(cs, kCTRL, rowbase, kk * 32, lane);
  }

  // ---- GRU: 8 column tiles of 16 covering CTRL=128 ----
#pragma unroll 1
  for (int c = 0; c < 8; ++c) {
    v8f gir = {}, ghr = {}, giz = {}, ghz = {}, gin = {}, ghn = {};
#pragma unroll
    for (int kk = 0; kk < 4; ++kk) {
      const int kb = kk * 32;
      gir = wmma_f16(aC[kk], loadB_T(W_ih, kCTRL, c * 16,       384, kb, lane), gir);
      giz = wmma_f16(aC[kk], loadB_T(W_ih, kCTRL, 128 + c * 16, 384, kb, lane), giz);
      gin = wmma_f16(aC[kk], loadB_T(W_ih, kCTRL, 256 + c * 16, 384, kb, lane), gin);
      ghr = wmma_f16(aH[kk], loadB_T(W_hh, kCTRL, c * 16,       384, kb, lane), ghr);
      ghz = wmma_f16(aH[kk], loadB_T(W_hh, kCTRL, 128 + c * 16, 384, kb, lane), ghz);
      ghn = wmma_f16(aH[kk], loadB_T(W_hh, kCTRL, 256 + c * 16, 384, kb, lane), ghn);
    }
    const int col = c * 16 + n;
    const float bir = b_ih[col],       bhr = b_hh[col];
    const float biz = b_ih[128 + col], bhz = b_hh[128 + col];
    const float bin = b_ih[256 + col], bhn = b_hh[256 + col];
#pragma unroll
    for (int r = 0; r < 8; ++r) {
      const int m = r + hi * 8;
      const int row = rowbase + m;
      const float hp = cs[(size_t)row * kCTRL + col];
      const float rg = sigmoidf_(gir[r] + bir + ghr[r] + bhr);
      const float zg = sigmoidf_(giz[r] + biz + ghz[r] + bhz);
      const float ng = tanhf(gin[r] + bin + rg * (ghn[r] + bhn));
      const float hn = (1.f - zg) * ng + zg * hp;
      o_h[(size_t)row * kCTRL + col] = hn;
      sH[m * kCTRL + col] = (_Float16)hn;
    }
  }
  __syncthreads();

  // ---- params = clip(h @ Wc^T + bc, +-20); TOTAL=268 -> 17 tiles ----
  v16h aP[4];
#pragma unroll
  for (int kk = 0; kk < 4; ++kk) aP[kk] = loadA_lds(sH, kk * 32, lane);

#pragma unroll 1
  for (int t = 0; t < 17; ++t) {
    v8f acc = {};
#pragma unroll
    for (int kk = 0; kk < 4; ++kk)
      acc = wmma_f16(aP[kk], loadB_T(Wc, kCTRL, t * 16, kTOT, kk * 32, lane), acc);
    const int col = t * 16 + n;
    if (col < kTOT) {
      const float bcv = bc[col];
#pragma unroll
      for (int r = 0; r < 8; ++r) {
        const int row = rowbase + r + hi * 8;
        float p = acc[r] + bcv;
        p = fminf(fmaxf(p, -20.f), 20.f);
        if (col < 140) {                       // head params, NPH=70 per head
          const int head = (col < 70) ? 0 : 1;
          const int j = col - head * 70;
          if (j < kDIM) {
            wsK[((size_t)row * 2 + head) * kDIM + j] = tanhf(p);
          } else if (j == kDIM) {
            wsBeta[(size_t)row * 2 + head] = (log1pf(expf(p)) + 1.f) * 2.f;  // SHARP=2
          }                                    // j in 65..69 unused
        } else if (col < 204) {
          wsE[(size_t)row * kDIM + (col - 140)] = sigmoidf_(p);
        } else {
          wsA[(size_t)row * kDIM + (col - 204)] = tanhf(p);
        }
      }
    }
  }
}

// ---------------------------------------------------------------------------
// Kernel 2: one block (256 threads) per batch row. Stages M/key_M in LDS once,
// computes cosine addressing + softmax, then one coalesced streaming pass for
// M_new / key_M copy / sum_aggre_new / read_vector.
// ---------------------------------------------------------------------------
__global__ __launch_bounds__(256) void k_memory(
    const float* __restrict__ M, const float* __restrict__ keyM,
    const float* __restrict__ w_aggre, const float* __restrict__ sum_aggre,
    const float* __restrict__ wsK, const float* __restrict__ wsBeta,
    const float* __restrict__ wsE, const float* __restrict__ wsA,
    float* __restrict__ o_rv, float* __restrict__ o_wr, float* __restrict__ o_ww,
    float* __restrict__ o_Mn, float* __restrict__ o_kM, float* __restrict__ o_wa,
    float* __restrict__ o_sa) {
  __shared__ float sM[kMEM * kDIM];       // 16 KB
  __shared__ float sKM[kMEM * kDIM];      // 16 KB
  __shared__ float sK[2 * kDIM];
  __shared__ float sE[kDIM], sA[kDIM];
  __shared__ float sBeta[2], sKn[2], sSum[2];
  __shared__ float sInnM[2 * kMEM], sInnK[2 * kMEM], sNM[kMEM], sNK[kMEM];
  __shared__ float sW[2 * kMEM];
  __shared__ float sRV[kDIM];

  const int t = threadIdx.x;
  const int b = blockIdx.x;
  const size_t base = (size_t)b * (kMEM * kDIM);

  // Prime the sum_aggre stream (consumed in phase 3) while we compute the
  // addressing phases: global_prefetch_b8, one per 64B chunk of this row.
  __builtin_prefetch(sum_aggre + base + (size_t)t * 16, 0, 1);

  // phase 0: stage per-row head parameters
  if (t < 128)       sK[t] = wsK[(size_t)b * 128 + t];
  else if (t < 192)  sE[t - 128] = wsE[(size_t)b * kDIM + (t - 128)];
  else               sA[t - 192] = wsA[(size_t)b * kDIM + (t - 192)];
  if (t < 2)    sBeta[t] = wsBeta[(size_t)b * 2 + t];
  if (t < kDIM) sRV[t] = 0.f;
  __syncthreads();

  // phase 1: stage M/key_M tiles, partial inner products + norms (4 lanes/slot)
  {
    const int slot = t >> 2, q = t & 3;
    float iM0 = 0, iM1 = 0, iK0 = 0, iK1 = 0, nM = 0, nK = 0;
#pragma unroll
    for (int i = 0; i < 4; ++i) {
      const int d = q * 16 + i * 4;
      const float4 mv = *(const float4*)(M + base + slot * kDIM + d);
      const float4 kv = *(const float4*)(keyM + base + slot * kDIM + d);
      *(float4*)(sM + slot * kDIM + d) = mv;
      *(float4*)(sKM + slot * kDIM + d) = kv;
      const float4 k0 = *(const float4*)(sK + d);
      const float4 k1 = *(const float4*)(sK + kDIM + d);
      iM0 += mv.x * k0.x + mv.y * k0.y + mv.z * k0.z + mv.w * k0.w;
      iM1 += mv.x * k1.x + mv.y * k1.y + mv.z * k1.z + mv.w * k1.w;
      iK0 += kv.x * k0.x + kv.y * k0.y + kv.z * k0.z + kv.w * k0.w;
      iK1 += kv.x * k1.x + kv.y * k1.y + kv.z * k1.z + kv.w * k1.w;
      nM  += mv.x * mv.x + mv.y * mv.y + mv.z * mv.z + mv.w * mv.w;
      nK  += kv.x * kv.x + kv.y * kv.y + kv.z * kv.z + kv.w * kv.w;
    }
#pragma unroll
    for (int off = 1; off < 4; off <<= 1) {
      iM0 += __shfl_xor(iM0, off);
      iM1 += __shfl_xor(iM1, off);
      iK0 += __shfl_xor(iK0, off);
      iK1 += __shfl_xor(iK1, off);
      nM  += __shfl_xor(nM, off);
      nK  += __shfl_xor(nK, off);
    }
    if (q == 0) {
      sInnM[slot] = iM0; sInnM[kMEM + slot] = iM1;
      sInnK[slot] = iK0; sInnK[kMEM + slot] = iK1;
      sNM[slot] = nM;    sNK[slot] = nK;
    }
  }
  __syncthreads();

  if (t < 2) {  // ||k_h||
    float s = 0.f;
    for (int d = 0; d < kDIM; ++d) { const float v = sK[t * kDIM + d]; s += v * v; }
    sKn[t] = sqrtf(s);
  }
  __syncthreads();

  // phase 2: K = 0.5*(cos(key_M,k)+cos(M,k)); amp = exp(beta*K)
  if (t < kMEM) {
    const float mn  = sqrtf(sNM[t]);
    const float kmn = sqrtf(sNK[t]);
#pragma unroll
    for (int h = 0; h < 2; ++h) {
      const float knh = sKn[h];
      const float cK = sInnK[h * kMEM + t] / (kmn * knh + 1e-8f);
      const float cM = sInnM[h * kMEM + t] / (mn  * knh + 1e-8f);
      sW[h * kMEM + t] = expf(sBeta[h] * 0.5f * (cK + cM));   // beta<=42 -> finite
    }
  }
  __syncthreads();
  if (t < 2) {
    float s = 0.f;
    for (int i = 0; i < kMEM; ++i) s += sW[t * kMEM + i];
    sSum[t] = s;
  }
  __syncthreads();
  if (t < kMEM) {
    const float wr = sW[t] / sSum[0];
    const float ww = sW[kMEM + t] / sSum[1];
    sW[t] = wr; sW[kMEM + t] = ww;
    o_wr[(size_t)b * kMEM + t] = wr;
    o_ww[(size_t)b * kMEM + t] = ww;
    o_wa[(size_t)b * kMEM + t] = w_aggre[(size_t)b * kMEM + t] + ww;
  }
  __syncthreads();

  // phase 3: streaming update (coalesced: 64 consecutive d per thread group)
  {
    const int d = t & 63;
    const int g = t >> 6;            // 4 groups of 16 slots
    const float ev = sE[d], av = sA[d];
    float rvAcc = 0.f;
#pragma unroll
    for (int mi = 0; mi < 16; ++mi) {
      const int m = g * 16 + mi;
      const float ww = sW[kMEM + m];
      const float wr = sW[m];
      const float Mv = sM[m * kDIM + d];
      const size_t gi = base + (size_t)m * kDIM + d;
      o_Mn[gi] = Mv * (1.f - ww * ev) + ww * av;
      o_sa[gi] = sum_aggre[gi] + ww * av;
      o_kM[gi] = sKM[m * kDIM + d];
      rvAcc += wr * Mv;
    }
    atomicAdd(&sRV[d], rvAcc);
  }
  __syncthreads();
  if (t < kDIM) o_rv[(size_t)b * kDIM + t] = sRV[t];
}

// ---------------------------------------------------------------------------
extern "C" void kernel_launch(void* const* d_in, const int* in_sizes, int n_in,
                              void* d_out, int out_size, void* d_ws, size_t ws_size,
                              hipStream_t stream) {
  const float* x         = (const float*)d_in[0];
  const float* prev_read = (const float*)d_in[1];
  const float* cs        = (const float*)d_in[2];
  const float* M         = (const float*)d_in[3];
  const float* keyM      = (const float*)d_in[4];
  const float* w_aggre   = (const float*)d_in[5];
  const float* sum_aggre = (const float*)d_in[6];
  const float* W_ih      = (const float*)d_in[7];
  const float* W_hh      = (const float*)d_in[8];
  const float* b_ih      = (const float*)d_in[9];
  const float* b_hh      = (const float*)d_in[10];
  const float* Wc        = (const float*)d_in[11];
  const float* bc        = (const float*)d_in[12];

  const int Bv = in_sizes[0] / kEMB;   // 8192

  // outputs: h, read_vector, w_read, w_write, M_new, key_M, w_aggre_new, sum_aggre_new
  float* out = (float*)d_out;
  float* o_h  = out;
  float* o_rv = o_h  + (size_t)Bv * kCTRL;
  float* o_wr = o_rv + (size_t)Bv * kDIM;
  float* o_ww = o_wr + (size_t)Bv * kMEM;
  float* o_Mn = o_ww + (size_t)Bv * kMEM;
  float* o_kM = o_Mn + (size_t)Bv * kMEM * kDIM;
  float* o_wa = o_kM + (size_t)Bv * kMEM * kDIM;
  float* o_sa = o_wa + (size_t)Bv * kMEM;

  // workspace: k (B*2*64), beta (B*2), erase (B*64), add (B*64) floats,
  // then f16 weight copies (W_ih 384x128, W_hh 384x128, Wc 268x128).
  float* ws     = (float*)d_ws;
  float* wsK    = ws;
  float* wsBeta = wsK    + (size_t)Bv * 2 * kDIM;
  float* wsE    = wsBeta + (size_t)Bv * 2;
  float* wsA    = wsE    + (size_t)Bv * kDIM;
  _Float16* fW  = (_Float16*)(wsA + (size_t)Bv * kDIM);
  _Float16* fWih = fW;
  _Float16* fWhh = fWih + (size_t)384 * kCTRL;
  _Float16* fWc  = fWhh + (size_t)384 * kCTRL;

  k_prep<<<(384 * kCTRL + 255) / 256, 256, 0, stream>>>(W_ih, W_hh, Wc, fWih, fWhh, fWc);
  k_controller<<<Bv / 16, 32, 0, stream>>>(x, prev_read, cs, fWih, fWhh, b_ih, b_hh,
                                           fWc, bc, o_h, wsK, wsBeta, wsE, wsA);
  k_memory<<<Bv, 256, 0, stream>>>(M, keyM, w_aggre, sum_aggre, wsK, wsBeta, wsE, wsA,
                                   o_rv, o_wr, o_ww, o_Mn, o_kM, o_wa, o_sa);
}
